// CodeFormer_635655160478
// MI455X (gfx1250) — matrix operand order repaired and checked by reading, not compiled, not run on hardware
//
#include <hip/hip_runtime.h>
#include <math.h>

// ---------------------------------------------------------------------------
// CDNA5 (gfx1250) VQGAN/CodeFormer forward.
// All convs / GEMMs run on v_wmma_f32_16x16x32_bf16 (f32 accumulate).
// Weight tiles staged in LDS as bf16 in [n][k] order so each lane's WMMA
// B-fragment is two 16B LDS loads; A-fragments are gathered
// branchlessly (clamped address + select) so vmem pipelines, or via
// global_load_b128 fast paths in the plain GEMM. Elementwise/normalization
// ops are bandwidth-trivial. Workspace: 3 * 16,777,216 floats + aux (~201MB).
// ---------------------------------------------------------------------------

typedef __attribute__((ext_vector_type(16))) __bf16 v16bf;
typedef __attribute__((ext_vector_type(8)))  float  v8f;

#define CF_EPS 1e-5f

static inline int ceil_div(int a, int b) { return (a + b - 1) / b; }

// pack 8 consecutive floats (16B-aligned) into afrag[base..base+7]
static __device__ __forceinline__ void cf_pack8(const float* p, v16bf& af, int base)
{
    float4 q0 = *(const float4*)p;
    float4 q1 = *(const float4*)(p + 4);
    af[base + 0] = (__bf16)q0.x; af[base + 1] = (__bf16)q0.y;
    af[base + 2] = (__bf16)q0.z; af[base + 3] = (__bf16)q0.w;
    af[base + 4] = (__bf16)q1.x; af[base + 5] = (__bf16)q1.y;
    af[base + 6] = (__bf16)q1.z; af[base + 7] = (__bf16)q1.w;
}

static __device__ __forceinline__ int cf_clampi(int v, int lo, int hi)
{
    return min(max(v, lo), hi);
}

// ========================= WMMA conv (implicit GEMM) =======================
// out[b,co,oh,ow] = bias[co] + sum_{ci,kh,kw} w[co,ci,kh,kw]*x[b,ci,ih,iw]
// M = B*OH*OW, N = Cout, K = Cin*KH*KW. 8 waves/block share one N-tile, so
// the weight tile is staged once per block in LDS (bf16, [n][k] order).
__global__ void __launch_bounds__(256)
cf_conv2d_wmma(const float* __restrict__ x, const float* __restrict__ w,
               const float* __restrict__ bias, float* __restrict__ y,
               int B, int Cin, int H, int W, int Cout, int KH, int KW,
               int stride, int pad, int OH, int OW, int act)
{
    const int K = Cin * KH * KW;
    const int M = B * OH * OW;
    const int lane = threadIdx.x & 31;
    const int wave = threadIdx.x >> 5;
    const int lh   = lane >> 4;     // half of wave (WMMA layout)
    const int mrow = lane & 15;
    const int n0 = blockIdx.x * 16;
    const int m0 = (blockIdx.y * 8 + wave) * 16;
    const bool mvalid = (m0 < M);

    __shared__ __attribute__((aligned(64))) __bf16 Bsh[16 * 32];   // [n][k]

    int b_ = 0, oh_ = 0, ow_ = 0;
    if (mvalid) {
        int m = m0 + mrow;
        ow_ = m % OW; int t = m / OW; oh_ = t % OH; b_ = t / OH;
    }

    v8f acc = {};
    for (int k0 = 0; k0 < K; k0 += 32) {
        // stage 16(N) x 32(K) weight tile -> LDS (bf16); branchless clamp+select
        for (int e = threadIdx.x; e < 512; e += 256) {
            int kk = e & 31, nn = e >> 5;
            int kg = k0 + kk, ng = n0 + nn;
            int kgc = cf_clampi(kg, 0, K - 1);
            int ngc = cf_clampi(ng, 0, Cout - 1);
            float v = w[(size_t)ngc * K + kgc];
            __builtin_prefetch(&w[(size_t)ngc * K + kgc + 32], 0, 1);
            v = (kg < K && ng < Cout) ? v : 0.f;
            Bsh[nn * 32 + kk] = (__bf16)v;
        }
        __syncthreads();

        // B 32x16 fragment: k = (j<8?j:j+8)+8*lh, n = mrow -> two 16B runs
        typedef __attribute__((ext_vector_type(8))) __bf16 v8bf_;
        v16bf bfrag = __builtin_shufflevector(*(const v8bf_*)(Bsh + mrow * 32 + 8 * lh), *(const v8bf_*)(Bsh + mrow * 32 + 16 + 8 * lh), 0,1,2,3,4,5,6,7,8,9,10,11,12,13,14,15);

        if (mvalid) {                       // wave-uniform: EXEC stays all-1s
            v16bf afrag;                    // A 16x32: k = (j<8?j:j+8)+8*lh
            #pragma unroll
            for (int j = 0; j < 16; ++j) {
                int kk = ((j < 8) ? j : (j + 8)) + 8 * lh;
                int kg = k0 + kk;
                int kgc = cf_clampi(kg, 0, K - 1);
                int kw_ = kgc % KW; int t2 = kgc / KW;
                int kh_ = t2 % KH; int ci = t2 / KH;
                int ih = oh_ * stride - pad + kh_;
                int iw = ow_ * stride - pad + kw_;
                bool inb = (kg < K) & (ih >= 0) & (ih < H) & (iw >= 0) & (iw < W);
                int ihc = cf_clampi(ih, 0, H - 1);
                int iwc = cf_clampi(iw, 0, W - 1);
                float v = x[(((size_t)b_ * Cin + ci) * H + ihc) * W + iwc];
                afrag[j] = (__bf16)(inb ? v : 0.f);
            }
            acc = __builtin_amdgcn_wmma_f32_16x16x32_bf16(
                    false, afrag, false, bfrag, (short)0, acc, false, false);
        }
        __syncthreads();
    }

    if (mvalid) {
        int n = n0 + mrow;
        if (n < Cout) {
            float bv = bias ? bias[n] : 0.f;
            #pragma unroll
            for (int r = 0; r < 8; ++r) {
                int m = m0 + r + 8 * lh;
                float v = acc[r] + bv;
                if (act == 1) v = fmaxf(v, 0.f);
                else if (act == 2) v = tanhf(v);
                int ow2 = m % OW; int t = m / OW; int oh2 = t % OH; int b2 = t / OH;
                y[(((size_t)b2 * Cout + n) * OH + oh2) * OW + ow2] = v;
            }
        }
    }
}

// ================= WMMA ConvTranspose2d (k=4, s=2, p=1) ====================
// out[b,co,oh,ow] = bias[co] + sum_{ci,kh,kw} w[ci,co,kh,kw]*x[b,ci,ih,iw]
// with ih = (oh + 1 - kh)/2 when integral & in range. Weight (Cin,Cout,4,4).
__global__ void __launch_bounds__(256)
cf_convT2d_wmma(const float* __restrict__ x, const float* __restrict__ w,
                const float* __restrict__ bias, float* __restrict__ y,
                int B, int Cin, int H, int W, int Cout, int OH, int OW)
{
    const int K = Cin * 16;
    const int M = B * OH * OW;
    const int lane = threadIdx.x & 31;
    const int wave = threadIdx.x >> 5;
    const int lh   = lane >> 4;
    const int mrow = lane & 15;
    const int n0 = blockIdx.x * 16;
    const int m0 = (blockIdx.y * 8 + wave) * 16;
    const bool mvalid = (m0 < M);

    __shared__ __attribute__((aligned(64))) __bf16 Bsh[16 * 32];   // [n][k]

    int b_ = 0, oh_ = 0, ow_ = 0;
    if (mvalid) {
        int m = m0 + mrow;
        ow_ = m % OW; int t = m / OW; oh_ = t % OH; b_ = t / OH;
    }

    v8f acc = {};
    for (int k0 = 0; k0 < K; k0 += 32) {
        for (int e = threadIdx.x; e < 512; e += 256) {
            int kk = e & 31, nn = e >> 5;
            int kg = k0 + kk, ng = n0 + nn;
            int kgc = cf_clampi(kg, 0, K - 1);
            int ngc = cf_clampi(ng, 0, Cout - 1);
            int ci = kgc >> 4, r = kgc & 15;
            float v = w[(((size_t)ci * Cout + ngc) << 4) + r];
            v = (kg < K && ng < Cout) ? v : 0.f;
            Bsh[nn * 32 + kk] = (__bf16)v;
        }
        __syncthreads();

        typedef __attribute__((ext_vector_type(8))) __bf16 v8bf_;
        v16bf bfrag = __builtin_shufflevector(*(const v8bf_*)(Bsh + mrow * 32 + 8 * lh), *(const v8bf_*)(Bsh + mrow * 32 + 16 + 8 * lh), 0,1,2,3,4,5,6,7,8,9,10,11,12,13,14,15);

        if (mvalid) {
            v16bf afrag;
            #pragma unroll
            for (int j = 0; j < 16; ++j) {
                int kk = ((j < 8) ? j : (j + 8)) + 8 * lh;
                int kg = k0 + kk;
                int kgc = cf_clampi(kg, 0, K - 1);
                int ci = kgc >> 4, kh = (kgc >> 2) & 3, kw = kgc & 3;
                int th = oh_ + 1 - kh, tw = ow_ + 1 - kw;
                int ih = th >> 1, iw = tw >> 1;   // arithmetic shift; clamped below
                bool inb = (kg < K) & (th >= 0) & (tw >= 0) &
                           (!(th & 1)) & (!(tw & 1)) & (ih < H) & (iw < W);
                int ihc = cf_clampi(ih, 0, H - 1);
                int iwc = cf_clampi(iw, 0, W - 1);
                float v = x[(((size_t)b_ * Cin + ci) * H + ihc) * W + iwc];
                afrag[j] = (__bf16)(inb ? v : 0.f);
            }
            acc = __builtin_amdgcn_wmma_f32_16x16x32_bf16(
                    false, afrag, false, bfrag, (short)0, acc, false, false);
        }
        __syncthreads();
    }

    if (mvalid) {
        int n = n0 + mrow;
        if (n < Cout) {
            float bv = bias ? bias[n] : 0.f;
            #pragma unroll
            for (int r = 0; r < 8; ++r) {
                int m = m0 + r + 8 * lh;
                int ow2 = m % OW; int t = m / OW; int oh2 = t % OH; int b2 = t / OH;
                y[(((size_t)b2 * Cout + n) * OH + oh2) * OW + ow2] = acc[r] + bv;
            }
        }
    }
}

// =========================== generic WMMA GEMM =============================
// C[m,n] = alpha * sum_k A[m,k] * B(k,n) + bias[n]; A row-major (lda);
// B(k,n) = transB ? Bm[n*ldb+k] : Bm[k*ldb+n]; z-slices via sA/sB/sC strides.
__global__ void __launch_bounds__(256)
cf_gemm_wmma(const float* __restrict__ A, const float* __restrict__ Bm,
             const float* __restrict__ bias, float* __restrict__ Cc,
             int M, int N, int K, int lda, int ldb, int ldc, int transB,
             float alpha, long long sA, long long sB, long long sC, int act)
{
    A  += (long long)blockIdx.z * sA;
    Bm += (long long)blockIdx.z * sB;
    Cc += (long long)blockIdx.z * sC;

    const int lane = threadIdx.x & 31;
    const int wave = threadIdx.x >> 5;
    const int lh   = lane >> 4;
    const int mrow = lane & 15;
    const int n0 = blockIdx.x * 16;
    const int m0 = (blockIdx.y * 8 + wave) * 16;
    const bool mvalid = (m0 < M);

    __shared__ __attribute__((aligned(64))) __bf16 Bsh[16 * 32];   // [n][k]

    v8f acc = {};
    for (int k0 = 0; k0 < K; k0 += 32) {
        for (int e = threadIdx.x; e < 512; e += 256) {
            int kk = e & 31, nn = e >> 5;
            int kg = k0 + kk, ng = n0 + nn;
            int kgc = cf_clampi(kg, 0, K - 1);
            int ngc = cf_clampi(ng, 0, N - 1);
            float v = transB ? Bm[(size_t)ngc * ldb + kgc]
                             : Bm[(size_t)kgc * ldb + ngc];
            v = (kg < K && ng < N) ? v : 0.f;
            Bsh[nn * 32 + kk] = (__bf16)v;
        }
        __syncthreads();

        typedef __attribute__((ext_vector_type(8))) __bf16 v8bf_;
        v16bf bfrag = __builtin_shufflevector(*(const v8bf_*)(Bsh + mrow * 32 + 8 * lh), *(const v8bf_*)(Bsh + mrow * 32 + 16 + 8 * lh), 0,1,2,3,4,5,6,7,8,9,10,11,12,13,14,15);

        if (mvalid) {
            int m = m0 + mrow;
            v16bf afrag;
            if (k0 + 32 <= K) {
                // two runs of 8 consecutive K values per lane -> 128-bit loads
                const float* pa = A + (size_t)m * lda + k0 + 8 * lh;
                cf_pack8(pa, afrag, 0);
                cf_pack8(pa + 16, afrag, 8);
            } else {
                #pragma unroll
                for (int j = 0; j < 16; ++j) {
                    int kk = ((j < 8) ? j : (j + 8)) + 8 * lh;
                    int kg = k0 + kk;
                    int kgc = cf_clampi(kg, 0, K - 1);
                    float v = A[(size_t)m * lda + kgc];
                    afrag[j] = (__bf16)((kg < K) ? v : 0.f);
                }
            }
            acc = __builtin_amdgcn_wmma_f32_16x16x32_bf16(
                    false, afrag, false, bfrag, (short)0, acc, false, false);
        }
        __syncthreads();
    }

    if (mvalid) {
        int n = n0 + mrow;
        if (n < N) {
            float bv = bias ? bias[n] : 0.f;
            #pragma unroll
            for (int r = 0; r < 8; ++r) {
                int m = m0 + r + 8 * lh;
                if (m < M) {
                    float v = alpha * acc[r] + bv;
                    if (act == 1) v = fmaxf(v, 0.f);
                    Cc[(size_t)m * ldc + n] = v;
                }
            }
        }
    }
}

// ============================ normalizations ===============================
__global__ void __launch_bounds__(256)
cf_groupnorm(const float* __restrict__ x, float* __restrict__ y,
             const float* __restrict__ gs, const float* __restrict__ gb,
             int C, int HW, int groups, int act)
{
    int bg = blockIdx.x;
    int g = bg % groups, bi = bg / groups;
    int cpg = C / groups;
    size_t base = ((size_t)bi * C + (size_t)g * cpg) * HW;
    size_t n = (size_t)cpg * HW;
    float s = 0.f, sq = 0.f;
    for (size_t i = threadIdx.x; i < n; i += 256) {
        float v = x[base + i]; s += v; sq += v * v;
    }
    __shared__ float rs[256], rq[256];
    rs[threadIdx.x] = s; rq[threadIdx.x] = sq;
    __syncthreads();
    for (int o = 128; o > 0; o >>= 1) {
        if ((int)threadIdx.x < o) {
            rs[threadIdx.x] += rs[threadIdx.x + o];
            rq[threadIdx.x] += rq[threadIdx.x + o];
        }
        __syncthreads();
    }
    float mean = rs[0] / (float)n;
    float var  = rq[0] / (float)n - mean * mean;
    float rstd = rsqrtf(var + CF_EPS);
    for (size_t i = threadIdx.x; i < n; i += 256) {
        int c = g * cpg + (int)(i / HW);
        float v = (x[base + i] - mean) * rstd * gs[c] + gb[c];
        if (act) v = fmaxf(v, 0.f);
        y[base + i] = v;
    }
}

__global__ void __launch_bounds__(256)
cf_layernorm(const float* __restrict__ x, float* __restrict__ y,
             const float* __restrict__ s, const float* __restrict__ b, int C)
{
    size_t row = blockIdx.x;
    const float* p = x + row * (size_t)C;
    float*       q = y + row * (size_t)C;
    float su = 0.f, sq = 0.f;
    for (int j = threadIdx.x; j < C; j += 256) { float v = p[j]; su += v; sq += v * v; }
    __shared__ float rs[256], rq[256];
    rs[threadIdx.x] = su; rq[threadIdx.x] = sq;
    __syncthreads();
    for (int o = 128; o > 0; o >>= 1) {
        if ((int)threadIdx.x < o) {
            rs[threadIdx.x] += rs[threadIdx.x + o];
            rq[threadIdx.x] += rq[threadIdx.x + o];
        }
        __syncthreads();
    }
    float mean = rs[0] / (float)C;
    float var  = rq[0] / (float)C - mean * mean;
    float rstd = rsqrtf(var + CF_EPS);
    for (int j = threadIdx.x; j < C; j += 256)
        q[j] = (p[j] - mean) * rstd * s[j] + b[j];
}

__global__ void __launch_bounds__(256)
cf_softmax_rows(float* __restrict__ S, int ncol)
{
    size_t row = blockIdx.x;
    float* p = S + row * (size_t)ncol;
    __shared__ float red[256];
    float mx = -3.4e38f;
    for (int j = threadIdx.x; j < ncol; j += 256) mx = fmaxf(mx, p[j]);
    red[threadIdx.x] = mx; __syncthreads();
    for (int o = 128; o > 0; o >>= 1) {
        if ((int)threadIdx.x < o) red[threadIdx.x] = fmaxf(red[threadIdx.x], red[threadIdx.x + o]);
        __syncthreads();
    }
    mx = red[0]; __syncthreads();
    float sum = 0.f;
    for (int j = threadIdx.x; j < ncol; j += 256) {
        float e = expf(p[j] - mx); p[j] = e; sum += e;
    }
    red[threadIdx.x] = sum; __syncthreads();
    for (int o = 128; o > 0; o >>= 1) {
        if ((int)threadIdx.x < o) red[threadIdx.x] += red[threadIdx.x + o];
        __syncthreads();
    }
    float inv = 1.f / red[0];
    for (int j = threadIdx.x; j < ncol; j += 256) p[j] *= inv;
}

// ========================= elementwise / transpose =========================
__global__ void cf_add_act(const float* __restrict__ a, const float* __restrict__ b,
                           float* __restrict__ y, size_t n, int act)
{
    size_t i = (size_t)blockIdx.x * blockDim.x + threadIdx.x;
    if (i < n) {
        float v = a[i] + b[i];
        if (act) v = fmaxf(v, 0.f);
        y[i] = v;
    }
}

__global__ void cf_nchw2nhwc(const float* __restrict__ x, float* __restrict__ y,
                             int B, int C, int HW)
{
    size_t i = (size_t)blockIdx.x * blockDim.x + threadIdx.x;
    size_t n = (size_t)B * C * HW;
    if (i < n) {
        int p = (int)(i % HW); size_t t = i / HW;
        int c = (int)(t % C);  int b = (int)(t / C);
        y[((size_t)b * HW + p) * C + c] = x[i];
    }
}

__global__ void cf_nhwc2nchw(const float* __restrict__ x, float* __restrict__ y,
                             int B, int C, int HW)
{
    size_t i = (size_t)blockIdx.x * blockDim.x + threadIdx.x;
    size_t n = (size_t)B * HW * C;
    if (i < n) {
        int c = (int)(i % C); size_t t = i / C;
        int p = (int)(t % HW); int b = (int)(t / HW);
        y[((size_t)b * C + c) * HW + p] = x[i];
    }
}

__global__ void cf_zero(float* __restrict__ p, int n)
{
    int i = blockIdx.x * 256 + threadIdx.x;
    if (i < n) p[i] = 0.f;
}

// ================================= VQ ======================================
__global__ void cf_cnorm(const float* __restrict__ cb, float* __restrict__ cn,
                         int Nc, int Cd)
{
    int j = blockIdx.x * 256 + threadIdx.x;
    if (j < Nc) {
        float s = 0.f;
        for (int k = 0; k < Cd; ++k) { float v = cb[(size_t)j * Cd + k]; s += v * v; }
        cn[j] = s;
    }
}

__global__ void __launch_bounds__(256)
cf_vq_argmin(const float* __restrict__ dots, const float* __restrict__ cn,
             const float* __restrict__ flat, const float* __restrict__ cb,
             float* __restrict__ quant, float* __restrict__ counts,
             float* __restrict__ sse, int Nc, int Cd)
{
    int m = blockIdx.x;
    float best = 3.4e38f; int bi = Nc;
    for (int j = threadIdx.x; j < Nc; j += 256) {
        float d = cn[j] - 2.f * dots[(size_t)m * Nc + j];
        if (d < best || (d == best && j < bi)) { best = d; bi = j; }
    }
    __shared__ float rv[256]; __shared__ int ri[256];
    rv[threadIdx.x] = best; ri[threadIdx.x] = bi;
    __syncthreads();
    for (int o = 128; o > 0; o >>= 1) {
        if ((int)threadIdx.x < o) {
            float ov = rv[threadIdx.x + o]; int oi = ri[threadIdx.x + o];
            if (ov < rv[threadIdx.x] || (ov == rv[threadIdx.x] && oi < ri[threadIdx.x])) {
                rv[threadIdx.x] = ov; ri[threadIdx.x] = oi;
            }
        }
        __syncthreads();
    }
    int idx = ri[0];
    __syncthreads();
    float local = 0.f;
    for (int k = threadIdx.x; k < Cd; k += 256) {
        float qv = cb[(size_t)idx * Cd + k];
        quant[(size_t)m * Cd + k] = qv;
        float d = qv - flat[(size_t)m * Cd + k];
        local += d * d;
    }
    rv[threadIdx.x] = local;
    __syncthreads();
    for (int o = 128; o > 0; o >>= 1) {
        if ((int)threadIdx.x < o) rv[threadIdx.x] += rv[threadIdx.x + o];
        __syncthreads();
    }
    if (threadIdx.x == 0) {
        atomicAdd(sse, rv[0]);
        atomicAdd(&counts[idx], 1.f);
    }
}

__global__ void __launch_bounds__(256)
cf_vq_finalize(const float* __restrict__ counts, const float* __restrict__ sse,
               float* __restrict__ out2, int Nc, float invRows, float invElems)
{
    float h = 0.f;
    for (int j = threadIdx.x; j < Nc; j += 256) {
        float p = counts[j] * invRows;
        h += p * logf(p + 1e-10f);
    }
    __shared__ float r[256];
    r[threadIdx.x] = h; __syncthreads();
    for (int o = 128; o > 0; o >>= 1) {
        if ((int)threadIdx.x < o) r[threadIdx.x] += r[threadIdx.x + o];
        __syncthreads();
    }
    if (threadIdx.x == 0) {
        out2[0] = 1.25f * sse[0] * invElems;  // q_loss + 0.25*e_loss (same MSE fwd)
        out2[1] = expf(-r[0]);                // perplexity
    }
}

// =========================== host-side helpers =============================
struct CfRB { const float *c1w,*c1b,*c2w,*c2b,*g1s,*g1b,*g2s,*g2b; };
struct CfAT { const float *lns,*lnb,*qw,*qb,*pw,*pb; };

static void launch_conv(hipStream_t s, const float* x, const float* w, const float* b,
                        float* y, int B, int Cin, int H, int W, int Cout,
                        int KH, int KW, int stride, int pad, int act)
{
    int OH = (H + 2 * pad - KH) / stride + 1;
    int OW = (W + 2 * pad - KW) / stride + 1;
    int M = B * OH * OW;
    dim3 g(ceil_div(Cout, 16), ceil_div(M, 128));
    cf_conv2d_wmma<<<g, 256, 0, s>>>(x, w, b, y, B, Cin, H, W, Cout, KH, KW,
                                     stride, pad, OH, OW, act);
}

static void launch_convT(hipStream_t s, const float* x, const float* w, const float* b,
                         float* y, int B, int Cin, int H, int W, int Cout)
{
    int OH = 2 * H, OW = 2 * W;
    int M = B * OH * OW;
    dim3 g(ceil_div(Cout, 16), ceil_div(M, 128));
    cf_convT2d_wmma<<<g, 256, 0, s>>>(x, w, b, y, B, Cin, H, W, Cout, OH, OW);
}

static void launch_gemm(hipStream_t s, const float* A, const float* Bm, const float* bias,
                        float* C, int M, int N, int K, int lda, int ldb, int ldc,
                        int transB, float alpha, long long sA, long long sB, long long sC,
                        int Z, int act)
{
    dim3 g(ceil_div(N, 16), ceil_div(M, 128), Z);
    cf_gemm_wmma<<<g, 256, 0, s>>>(A, Bm, bias, C, M, N, K, lda, ldb, ldc,
                                   transB, alpha, sA, sB, sC, act);
}

static void launch_gn(hipStream_t s, const float* x, float* y, const float* gs,
                      const float* gb, int Bn, int C, int HW, int act)
{
    cf_groupnorm<<<Bn * 32, 256, 0, s>>>(x, y, gs, gb, C, HW, 32, act);
}

static void run_resblock(hipStream_t s, float* X, float* T1, float* T2,
                         const CfRB& p, int Bn, int C, int Hs)
{
    size_t n = (size_t)Bn * C * Hs * Hs;
    launch_conv(s, X, p.c1w, p.c1b, T1, Bn, C, Hs, Hs, C, 3, 3, 1, 1, 0);
    launch_gn(s, T1, T1, p.g1s, p.g1b, Bn, C, Hs * Hs, 1);
    launch_conv(s, T1, p.c2w, p.c2b, T2, Bn, C, Hs, Hs, C, 3, 3, 1, 1, 0);
    launch_gn(s, T2, T2, p.g2s, p.g2b, Bn, C, Hs * Hs, 0);
    cf_add_act<<<ceil_div((int)n, 256), 256, 0, s>>>(T2, X, X, n, 1);
}

static void run_attn(hipStream_t s, float* X, float* T1, float* T2,
                     const CfAT& p, int Bn, int C, int Hs)
{
    const int N = Hs * Hs, heads = 8, hd = C / heads;
    const size_t NC = (size_t)Bn * N * C;
    float* seq = T1;
    float* xn  = T1 + NC;
    float* qkv = T1 + 2 * NC;     // (Bn*N, 3C)
    float* O   = T1 + 5 * NC;
    float* P   = T1 + 6 * NC;
    float* S   = T2;              // (Bn, heads, N, N)
    float scale = 1.0f / sqrtf((float)hd);

    cf_nchw2nhwc<<<ceil_div((int)NC, 256), 256, 0, s>>>(X, seq, Bn, C, N);
    cf_layernorm<<<Bn * N, 256, 0, s>>>(seq, xn, p.lns, p.lnb, C);
    launch_gemm(s, xn, p.qw, p.qb, qkv, Bn * N, 3 * C, C, C, 3 * C, 3 * C,
                0, 1.f, 0, 0, 0, 1, 0);
    for (int b = 0; b < Bn; ++b) {   // S = scale * q @ k^T, z over heads
        const float* q = qkv + (size_t)b * N * 3 * C;
        const float* k = q + C;
        float* Sb = S + (size_t)b * heads * N * N;
        launch_gemm(s, q, k, nullptr, Sb, N, N, hd, 3 * C, 3 * C, N,
                    1, scale, hd, hd, (long long)N * N, heads, 0);
    }
    cf_softmax_rows<<<Bn * heads * N, 256, 0, s>>>(S, N);
    for (int b = 0; b < Bn; ++b) {   // O = softmax(S) @ v, written (N, heads*hd)
        const float* Sb = S + (size_t)b * heads * N * N;
        const float* v = qkv + (size_t)b * N * 3 * C + 2 * C;
        float* Ob = O + (size_t)b * N * C;
        launch_gemm(s, Sb, v, nullptr, Ob, N, hd, N, N, 3 * C, C,
                    0, 1.f, (long long)N * N, hd, hd, heads, 0);
    }
    launch_gemm(s, O, p.pw, p.pb, P, Bn * N, C, C, C, C, C, 0, 1.f, 0, 0, 0, 1, 0);
    cf_add_act<<<ceil_div((int)NC, 256), 256, 0, s>>>(P, seq, seq, NC, 0);
    cf_nhwc2nchw<<<ceil_div((int)NC, 256), 256, 0, s>>>(seq, X, Bn, C, N);
}

static void run_vq(hipStream_t s, const float* feat, float* scr, float* qout,
                   const float* cb, float* aux, float* out_tail)
{
    const int Bn = 2, Cd = 256, HW = 1024, Nc = 1024, Rows = 2048;
    float* flat  = scr;                       // 2048*256
    float* dots  = scr + ((size_t)1 << 21);   // 2048*1024
    float* quant = scr + ((size_t)1 << 23);   // 2048*256
    float* cn = aux; float* counts = aux + 1024; float* sse = aux + 2048;

    cf_nchw2nhwc<<<ceil_div(Bn * Cd * HW, 256), 256, 0, s>>>(feat, flat, Bn, Cd, HW);
    cf_cnorm<<<ceil_div(Nc, 256), 256, 0, s>>>(cb, cn, Nc, Cd);
    cf_zero<<<ceil_div(1025, 256), 256, 0, s>>>(counts, 1025);   // counts + sse
    launch_gemm(s, flat, cb, nullptr, dots, Rows, Nc, Cd, Cd, Cd, Nc,
                1, 1.f, 0, 0, 0, 1, 0);
    cf_vq_argmin<<<Rows, 256, 0, s>>>(dots, cn, flat, cb, quant, counts, sse, Nc, Cd);
    cf_vq_finalize<<<1, 256, 0, s>>>(counts, sse, out_tail, Nc,
                                     1.f / (float)Rows, 1.f / ((float)Rows * Cd));
    cf_nhwc2nchw<<<ceil_div(Bn * Cd * HW, 256), 256, 0, s>>>(quant, qout, Bn, Cd, HW);
}

// ================================ driver ===================================
extern "C" void kernel_launch(void* const* d_in, const int* in_sizes, int n_in,
                              void* d_out, int out_size, void* d_ws, size_t ws_size,
                              hipStream_t stream)
{
    (void)in_sizes; (void)n_in; (void)out_size; (void)ws_size;

    int ci = 0;
    auto nx = [&]() { return (const float*)d_in[ci++]; };
    auto rdRB = [&]() {
        CfRB r; r.c1w = nx(); r.c1b = nx(); r.c2w = nx(); r.c2b = nx();
        r.g1s = nx(); r.g1b = nx(); r.g2s = nx(); r.g2b = nx(); return r;
    };
    auto rdAT = [&]() {
        CfAT a; a.lns = nx(); a.lnb = nx(); a.qw = nx(); a.qb = nx();
        a.pw = nx(); a.pb = nx(); return a;
    };

    // ---- leaves in setup_inputs() insertion order ----
    const float* xin = nx();
    const float *e_cin_w = nx(), *e_cin_b = nx();
    CfRB e1a = rdRB(), e1b = rdRB(); const float *e1dw = nx(), *e1db = nx();
    CfRB e2a = rdRB(), e2b = rdRB(); const float *e2dw = nx(), *e2db = nx();
    CfRB e3a = rdRB(), e3b = rdRB(); const float *e3dw = nx(), *e3db = nx();
    CfRB em1 = rdRB(); CfAT emat = rdAT(); CfRB em2 = rdRB();
    const float *e_co_w = nx(), *e_co_b = nx();
    const float *d_cin_w = nx(), *d_cin_b = nx();
    CfRB dm1 = rdRB(); CfAT dmat = rdAT(); CfRB dm2 = rdRB();
    CfRB u1a = rdRB(), u1b = rdRB(); const float *u1w = nx(), *u1bs = nx();
    CfRB u2a = rdRB(), u2b = rdRB(); const float *u2w = nx(), *u2bs = nx();
    CfRB u3a = rdRB(), u3b = rdRB(); const float *u3w = nx(), *u3bs = nx();
    const float *d_co_w = nx(), *d_co_b = nx();
    const float *codebook = nx();

    // ---- workspace: 3 ping-pong buffers + aux ----
    const size_t BIG = (size_t)2 * 128 * 256 * 256;   // 16,777,216 floats
    float* X   = (float*)d_ws;
    float* T1  = X + BIG;
    float* T2  = X + 2 * BIG;
    float* AUX = X + 3 * BIG;
    float* out = (float*)d_out;
    hipStream_t s = stream;
    auto swapXT1 = [&]() { float* t = X; X = T1; T1 = t; };

    // ================= encoder =================
    launch_conv(s, xin, e_cin_w, e_cin_b, X, 2, 3, 256, 256, 128, 3, 3, 1, 1, 0);
    run_resblock(s, X, T1, T2, e1a, 2, 128, 256);
    run_resblock(s, X, T1, T2, e1b, 2, 128, 256);
    launch_conv(s, X, e1dw, e1db, T1, 2, 128, 256, 256, 256, 4, 4, 2, 1, 0); swapXT1();
    run_resblock(s, X, T1, T2, e2a, 2, 256, 128);
    run_resblock(s, X, T1, T2, e2b, 2, 256, 128);
    launch_conv(s, X, e2dw, e2db, T1, 2, 256, 128, 128, 512, 4, 4, 2, 1, 0); swapXT1();
    run_resblock(s, X, T1, T2, e3a, 2, 512, 64);
    run_resblock(s, X, T1, T2, e3b, 2, 512, 64);
    launch_conv(s, X, e3dw, e3db, T1, 2, 512, 64, 64, 512, 4, 4, 2, 1, 0); swapXT1();
    run_resblock(s, X, T1, T2, em1, 2, 512, 32);
    run_attn(s, X, T1, T2, emat, 2, 512, 32);
    run_resblock(s, X, T1, T2, em2, 2, 512, 32);
    launch_conv(s, X, e_co_w, e_co_b, T1, 2, 512, 32, 32, 256, 1, 1, 1, 0, 0);  // feat

    // ================= VQ (loss + perplexity -> out[393216..]) ============
    run_vq(s, T1, T2, X, codebook, AUX, out + (size_t)2 * 3 * 256 * 256);

    // ================= decoder =================
    launch_conv(s, X, d_cin_w, d_cin_b, T1, 2, 256, 32, 32, 512, 1, 1, 1, 0, 0); swapXT1();
    run_resblock(s, X, T1, T2, dm1, 2, 512, 32);
    run_attn(s, X, T1, T2, dmat, 2, 512, 32);
    run_resblock(s, X, T1, T2, dm2, 2, 512, 32);
    run_resblock(s, X, T1, T2, u1a, 2, 512, 32);
    run_resblock(s, X, T1, T2, u1b, 2, 512, 32);
    launch_convT(s, X, u1w, u1bs, T1, 2, 512, 32, 32, 256); swapXT1();
    run_resblock(s, X, T1, T2, u2a, 2, 256, 64);
    run_resblock(s, X, T1, T2, u2b, 2, 256, 64);
    launch_convT(s, X, u2w, u2bs, T1, 2, 256, 64, 64, 128); swapXT1();
    run_resblock(s, X, T1, T2, u3a, 2, 128, 128);
    run_resblock(s, X, T1, T2, u3b, 2, 128, 128);
    launch_convT(s, X, u3w, u3bs, T1, 2, 128, 128, 128, 64); swapXT1();
    launch_conv(s, X, d_co_w, d_co_b, out, 2, 64, 256, 256, 3, 3, 3, 1, 1, 2); // tanh
}
